// SessionGraph_14955076124882
// MI455X (gfx1250) — matrix operand order repaired and checked by reading, not compiled, not run on hardware
//
#include <hip/hip_runtime.h>
#include <hip/hip_bf16.h>

typedef __attribute__((ext_vector_type(16))) _Float16 v16h;
typedef __attribute__((ext_vector_type(8)))  _Float16 v8h;
typedef __attribute__((ext_vector_type(8)))  float    v8f;

// ---------------------------------------------------------------------------
// f32 -> f16 conversion (weights, adjacency)
// ---------------------------------------------------------------------------
__global__ void cvt_f32_f16_kernel(const float* __restrict__ src,
                                   _Float16* __restrict__ dst, long long n) {
    long long i = (long long)blockIdx.x * blockDim.x + threadIdx.x;
    long long stride = (long long)gridDim.x * blockDim.x;
    for (; i < n; i += stride) dst[i] = (_Float16)src[i];
}

// ---------------------------------------------------------------------------
// Embedding gather + L2 normalize. One wave (32 lanes) per row of H=256.
// ---------------------------------------------------------------------------
__global__ void gather_norm_kernel(const int* __restrict__ idx,
                                   const float* __restrict__ emb,
                                   float* __restrict__ h32,
                                   _Float16* __restrict__ h16, int rows) {
    const int H = 256;
    int wpb  = blockDim.x >> 5;
    int row  = blockIdx.x * wpb + (threadIdx.x >> 5);
    if (row >= rows) return;
    int lane = threadIdx.x & 31;
    const float* src = emb + (long long)idx[row] * H;
    float v[8];
    float ss = 0.f;
#pragma unroll
    for (int j = 0; j < 8; ++j) { v[j] = src[lane * 8 + j]; ss += v[j] * v[j]; }
#pragma unroll
    for (int o = 16; o > 0; o >>= 1) ss += __shfl_xor(ss, o, 32);
    float inv = 1.0f / (sqrtf(ss) + 1e-12f);
    float*    d32 = h32 + (long long)row * H + lane * 8;
    _Float16* d16 = h16 + (long long)row * H + lane * 8;
#pragma unroll
    for (int j = 0; j < 8; ++j) {
        float x = v[j] * inv;
        d32[j] = x;
        d16[j] = (_Float16)x;
    }
}

// ---------------------------------------------------------------------------
// Batched WMMA GEMM: C[b][m][n] = sum_k A[b][m][k] * B(k,n) + bias[n]
// B operand is always per-lane contiguous: B(k,n) = Bmat[n*ldb + k]
// (i.e. X @ W^T with W stored [N,K] row-major, or a pre-transposed K-major
// activation tensor) -> fragments load as global_load_b128 pairs.
//
// One wave computes a 16x64 strip: 4 independent 16x16 accumulators share the
// A fragment per 32-wide K step (amortizes A traffic, pipelines the XDL).
// Fragment layouts per CDNA5 ISA 7.12.2:
//   A 16x32 f16: lane half h, row = lane&15; K in {8h..8h+7} U {16+8h..16+8h+7}
//   B 32x16 f16: col = lane&15; K in {8h..8h+7} U {16+8h..16+8h+7}
//   C f32: VGPR i -> row 8h+i, col lane&15
//
// CT=0: normal store   C16[b*strideC + row*ldc + col]
// CT=1: per-64-row-batch transposed store (k-major activations for the
//       downstream batched GEMM): C16[(row>>6)*strideC + col*ldc + (row&63)]
// ---------------------------------------------------------------------------
template <int CT>
__global__ void wmma_gemm_kernel(const _Float16* __restrict__ Amat,
                                 const _Float16* __restrict__ Bmat,
                                 const float* __restrict__ bias,
                                 _Float16* __restrict__ C16,
                                 int K, int lda, int ldb, int ldc,
                                 long long strideA, long long strideB, long long strideC,
                                 int tilesM, int tilesNq, long long totalWaves) {
    long long wave = (long long)blockIdx.x * (blockDim.x >> 5) + (threadIdx.x >> 5);
    if (wave >= totalWaves) return;              // wave-uniform; EXEC stays all-1s
    int lane = threadIdx.x & 31;
    int hgrp = lane >> 4;                        // lane half (0/1)
    int r    = lane & 15;

    int tilesPerBatch = tilesM * tilesNq;
    int batch = (int)(wave / tilesPerBatch);
    int t     = (int)(wave % tilesPerBatch);
    int tm = t / tilesNq;
    int tn = (t % tilesNq) * 4;                  // first of 4 N-tiles

    const _Float16* Ab = Amat + (long long)batch * strideA;
    const _Float16* Bb = Bmat + (long long)batch * strideB;

    int m = tm * 16 + r;                         // A row handled by this lane
    int n = tn * 16 + r;                         // first B column for this lane

    v8f acc0 = {}, acc1 = {}, acc2 = {}, acc3 = {};

    for (int k0 = 0; k0 < K; k0 += 32) {
        // ---- A fragment: two contiguous 16B chunks per lane ----
        const _Float16* ap = Ab + (long long)m * lda + k0 + 8 * hgrp;
        v8h a0 = *(const v8h*)(ap);              // K = 8h .. 8h+7
        v8h a1 = *(const v8h*)(ap + 16);         // K = 16+8h .. 16+8h+7
        v16h a = __builtin_shufflevector(a0, a1, 0, 1, 2, 3, 4, 5, 6, 7,
                                         8, 9, 10, 11, 12, 13, 14, 15);

        // ---- B fragments for the 4 N-tiles: two contiguous 16B chunks per lane ----
        v16h b[4];
#pragma unroll
        for (int j = 0; j < 4; ++j) {
            const _Float16* bp = Bb + (long long)(n + 16 * j) * ldb + k0 + 8 * hgrp;
            v8h b0 = *(const v8h*)(bp);
            v8h b1 = *(const v8h*)(bp + 16);
            b[j] = __builtin_shufflevector(b0, b1, 0, 1, 2, 3, 4, 5, 6, 7,
                                           8, 9, 10, 11, 12, 13, 14, 15);
        }

        acc0 = __builtin_amdgcn_wmma_f32_16x16x32_f16(false, a, false, b[0],
                                                      (short)0, acc0, false, false);
        acc1 = __builtin_amdgcn_wmma_f32_16x16x32_f16(false, a, false, b[1],
                                                      (short)0, acc1, false, false);
        acc2 = __builtin_amdgcn_wmma_f32_16x16x32_f16(false, a, false, b[2],
                                                      (short)0, acc2, false, false);
        acc3 = __builtin_amdgcn_wmma_f32_16x16x32_f16(false, a, false, b[3],
                                                      (short)0, acc3, false, false);
    }

    // ---- epilogue: bias + f16 store (optionally batch-transposed) ----
    v8f accs[4] = {acc0, acc1, acc2, acc3};
#pragma unroll
    for (int j = 0; j < 4; ++j) {
        float bv = bias[(tn + j) * 16 + r];
#pragma unroll
        for (int i = 0; i < 8; ++i) {
            int row = tm * 16 + 8 * hgrp + i;
            int col = (tn + j) * 16 + r;
            long long off;
            if (CT) {
                off = (long long)(row >> 6) * strideC + (long long)col * ldc + (row & 63);
            } else {
                off = (long long)batch * strideC + (long long)row * ldc + col;
            }
            C16[off] = (_Float16)(accs[j][i] + bv);
        }
    }
}

// ---------------------------------------------------------------------------
// GRU gate math: out = ng + ig * (h - ng)
// ---------------------------------------------------------------------------
__global__ void gates_kernel(const _Float16* __restrict__ gi,
                             const _Float16* __restrict__ gh,
                             const float* __restrict__ h32,
                             float* __restrict__ out, long long total) {
    long long i = (long long)blockIdx.x * blockDim.x + threadIdx.x;
    if (i >= total) return;
    long long row = i >> 8;          // H = 256
    int       col = (int)(i & 255);
    const _Float16* gir = gi + row * 768;
    const _Float16* ghr = gh + row * 768;
    float ir  = (float)gir[col]       + (float)ghr[col];
    float ii  = (float)gir[256 + col] + (float)ghr[256 + col];
    float in_ = (float)gir[512 + col];
    float hn  = (float)ghr[512 + col];
    float rg = 1.0f / (1.0f + __expf(-ir));
    float ig = 1.0f / (1.0f + __expf(-ii));
    float ng = tanhf(in_ + rg * hn);
    float hv = h32[i];
    out[i] = ng + ig * (hv - ng);
}

// ---------------------------------------------------------------------------
extern "C" void kernel_launch(void* const* d_in, const int* in_sizes, int n_in,
                              void* d_out, int out_size, void* d_ws, size_t ws_size,
                              hipStream_t stream) {
    const int B = 512, N = 64, H = 256;
    const int R = B * N;                         // 32768 rows

    const int*   inputs = (const int*)  d_in[0];
    const float* A      = (const float*)d_in[1];
    const float* emb    = (const float*)d_in[2];
    const float* W_in   = (const float*)d_in[3];
    const float* b_in   = (const float*)d_in[4];
    const float* W_out  = (const float*)d_in[5];
    const float* b_out  = (const float*)d_in[6];
    const float* w_ih   = (const float*)d_in[7];
    const float* b_ih   = (const float*)d_in[8];
    const float* w_hh   = (const float*)d_in[9];
    const float* b_hh   = (const float*)d_in[10];
    const float* b_iah  = (const float*)d_in[11];
    const float* b_oah  = (const float*)d_in[12];
    float* out = (float*)d_out;

    char* ws = (char*)d_ws;
    size_t off = 0;
    auto alloc = [&](size_t bytes) -> void* {
        void* p = ws + off;
        off += (bytes + 255) & ~(size_t)255;
        return p;
    };

    float*    h32    = (float*)   alloc((size_t)R * H * 4);        // 33.5 MB
    _Float16* h16    = (_Float16*)alloc((size_t)R * H * 2);        // 16.8 MB
    _Float16* hinT   = (_Float16*)alloc((size_t)R * H * 2);        // [B][H=256][64] k-major
    _Float16* houtT  = (_Float16*)alloc((size_t)R * H * 2);
    _Float16* inp    = (_Float16*)alloc((size_t)R * 2 * H * 2);    // 33.5 MB
    _Float16* gi     = (_Float16*)alloc((size_t)R * 3 * H * 2);    // 50.3 MB
    _Float16* gh     = (_Float16*)alloc((size_t)R * 3 * H * 2);    // 50.3 MB
    _Float16* A16    = (_Float16*)alloc((size_t)B * N * 2 * N * 2);// 8.4 MB
    _Float16* Win16  = (_Float16*)alloc((size_t)H * H * 2);
    _Float16* Wout16 = (_Float16*)alloc((size_t)H * H * 2);
    _Float16* wih16  = (_Float16*)alloc((size_t)3 * H * 2 * H * 2);
    _Float16* whh16  = (_Float16*)alloc((size_t)3 * H * H * 2);

    auto cvt = [&](const float* s, _Float16* d, long long n) {
        int blocks = (int)((n + 1023) / 1024);
        cvt_f32_f16_kernel<<<blocks, 256, 0, stream>>>(s, d, n);
    };
    cvt(W_in,  Win16,  (long long)H * H);
    cvt(W_out, Wout16, (long long)H * H);
    cvt(w_ih,  wih16,  (long long)3 * H * 2 * H);
    cvt(w_hh,  whh16,  (long long)3 * H * H);
    cvt(A,     A16,    (long long)B * N * 2 * N);

    // gather + L2 normalize: 8 waves per block, one wave per row
    gather_norm_kernel<<<R / 8, 256, 0, stream>>>(inputs, emb, h32, h16, R);

    auto gemm = [&](const _Float16* Am, const _Float16* Bm, const float* bias,
                    _Float16* C16, int M, int Nn, int K,
                    int lda, int ldb, int ldc,
                    long long sA, long long sB, long long sC,
                    int batches, int ct) {
        int tilesM  = M / 16;
        int tilesNq = Nn / 64;                   // 4 N-tiles per wave
        long long waves = (long long)batches * tilesM * tilesNq;
        int blocks = (int)((waves + 7) / 8);
        if (ct) {
            wmma_gemm_kernel<1><<<blocks, 256, 0, stream>>>(
                Am, Bm, bias, C16, K, lda, ldb, ldc, sA, sB, sC,
                tilesM, tilesNq, waves);
        } else {
            wmma_gemm_kernel<0><<<blocks, 256, 0, stream>>>(
                Am, Bm, bias, C16, K, lda, ldb, ldc, sA, sB, sC,
                tilesM, tilesNq, waves);
        }
    };

    // hinT  = (h @ W_in^T  + b_in )^T per batch : [B][256][64], k-major
    // houtT = (h @ W_out^T + b_out)^T per batch
    gemm(h16, Win16,  b_in,  hinT,  R, H, H, H, H, N,
         0, 0, (long long)H * N, 1, 1);
    gemm(h16, Wout16, b_out, houtT, R, H, H, H, H, N,
         0, 0, (long long)H * N, 1, 1);

    // input_in  = A[:, :, :64] @ hin  + b_iah -> inp[:, :256]   (ldc = 512)
    // input_out = A[:, :, 64:] @ hout + b_oah -> inp[:, 256:]
    // B operand is k-major hinT/houtT: B(k,n) = hinT[b][n*64 + k]  (ldb = 64)
    gemm(A16,     hinT,  b_iah, inp,     N, H, N,
         2 * N, N, 2 * H,
         (long long)N * 2 * N, (long long)H * N, (long long)N * 2 * H, B, 0);
    gemm(A16 + N, houtT, b_oah, inp + H, N, H, N,
         2 * N, N, 2 * H,
         (long long)N * 2 * N, (long long)H * N, (long long)N * 2 * H, B, 0);

    // gi = inp @ w_ih^T + b_ih   [32768 x 768], K = 512
    gemm(inp, wih16, b_ih, gi, R, 3 * H, 2 * H, 2 * H, 2 * H, 3 * H,
         0, 0, 0, 1, 0);
    // gh = h @ w_hh^T + b_hh     [32768 x 768], K = 256
    gemm(h16, whh16, b_hh, gh, R, 3 * H, H, H, H, 3 * H,
         0, 0, 0, 1, 0);

    // GRU gates -> output
    long long total = (long long)R * H;          // 8,388,608
    gates_kernel<<<(int)((total + 255) / 256), 256, 0, stream>>>(gi, gh, h32, out, total);
}
